// StructRefiner_66065186947187
// MI455X (gfx1250) — matrix operand order repaired and checked by reading, not compiled, not run on hardware
//
#include <hip/hip_runtime.h>

// ---------------- problem constants (match reference) ----------------
#define KN       64      // neighbors per anchor
#define DD       500     // complex half-dim
#define TWO_D    1000    // embedding width
#define ATTN_DIM 128
#define ETA_MAX  0.5f
#define INV_SQRT_ATTN 0.08838834764831844f  // 1/sqrt(128)

// LDS layout offsets (bytes) for refiner_main
#define OFF_EI    128000                    // after hat (64*1000*2)
#define OFF_V     132000
#define OFF_AT    136000
#define OFF_HAS   136256
#define OFF_STAGE 136272                    // 8 waves * 2 bufs * 4096 B
#define SMEM_TOTAL (OFF_STAGE + 8 * 2 * 4096)

// ---------------- WMMA types ----------------
typedef __attribute__((ext_vector_type(16))) __bf16 v16bf;
typedef __attribute__((ext_vector_type(8)))  float  v8f;

// ---------------- async global->LDS copy (gfx1250 path) ----------------
#if __has_builtin(__builtin_amdgcn_global_load_async_to_lds_b128)
#define HAVE_ASYNC_LDS 1
#else
#define HAVE_ASYNC_LDS 0
#endif

// builtin signature (from clang diagnostic): arg0 = AS1 pointer to
// int __attribute__((vector_size(16))), arg1 = AS3 pointer, then Imm offset, Imm cpol
typedef int v4i_t __attribute__((vector_size(16)));
typedef __attribute__((address_space(1))) v4i_t as1_v4i;
typedef __attribute__((address_space(3))) v4i_t as3_v4i;

__device__ __forceinline__ void cp16_g2l(const char* g, char* l) {
#if HAVE_ASYNC_LDS
    __builtin_amdgcn_global_load_async_to_lds_b128((as1_v4i*)g, (as3_v4i*)l, 0, 0);
#else
    *(float4*)l = *(const float4*)g;          // synchronous fallback
#endif
}

template <int N>
__device__ __forceinline__ void wait_async() {
#if HAVE_ASYNC_LDS
#if __has_builtin(__builtin_amdgcn_s_wait_asynccnt)
    __builtin_amdgcn_s_wait_asynccnt(N);
#else
    asm volatile("s_wait_asynccnt %0" :: "i"(N) : "memory");
#endif
#endif
}

// copy one 4000-byte embedding row (16B aligned) into an LDS buffer: 8 x b128/lane
__device__ __forceinline__ void issue_row(const float* grow, float* lbuf, int lane) {
#pragma unroll
    for (int i = 0; i < 8; ++i) {
        const int off = i * 512 + lane * 16;
        if (off < 4000)
            cp16_g2l((const char*)grow + off, (char*)lbuf + off);
    }
}

// =====================================================================
// GEMM 1: q[B,128] = gather(emb, anchor_ids) @ Wq^T    (bf16 WMMA)
// one wave per 16x16 tile; 31 unguarded K-iters + 1 clamped tail
// =====================================================================
__global__ void __launch_bounds__(32)
qproj_wmma(const int* __restrict__ anchor_ids,
           const float* __restrict__ emb,
           const float* __restrict__ Wq,
           float* __restrict__ q_ws)
{
    const int lane = threadIdx.x;
    const int m  = lane & 15;
    const int hh = lane >> 4;

    const float* arow = emb + (long long)anchor_ids[blockIdx.x * 16 + m] * TWO_D;
    const float* wrow = Wq  + (long long)(blockIdx.y * 16 + m) * TWO_D;
    const float2* arow2 = (const float2*)arow;   // rows are 8B aligned
    const float2* wrow2 = (const float2*)wrow;

    v8f c = {};
    for (int k0 = 0; k0 < 992; k0 += 32) {       // fully in-bounds: k0+31 <= 991
        v16bf a, b;
#pragma unroll
        for (int g = 0; g < 8; ++g) {
            const int kA = k0 + ((g >> 2) << 4) + (hh << 3) + ((g & 3) << 1);
            const float2 ta = arow2[kA >> 1];
            a[2*g]   = (__bf16)ta.x;             // hw v_cvt (pairs -> v_cvt_pk_bf16_f32)
            a[2*g+1] = (__bf16)ta.y;
            const int kB = k0 + ((g >> 2) << 4) + (hh << 3) + ((g & 3) << 1);
            const float2 tb = wrow2[kB >> 1];
            b[2*g]   = (__bf16)tb.x;
            b[2*g+1] = (__bf16)tb.y;
        }
        c = __builtin_amdgcn_wmma_f32_16x16x32_bf16(false, a, false, b,
                                                    (short)0, c, false, false);
    }
    {   // tail k0 = 992: clamp addresses, select values (branch-free)
        v16bf a, b;
#pragma unroll
        for (int g = 0; g < 8; ++g) {
            const int kA  = 992 + ((g >> 2) << 4) + (hh << 3) + ((g & 3) << 1);
            const int kAc = (kA < 998) ? kA : 998;
            const float2 ta = arow2[kAc >> 1];
            a[2*g]   = (__bf16)((kA     < TWO_D) ? ta.x : 0.0f);
            a[2*g+1] = (__bf16)((kA + 1 < TWO_D) ? ta.y : 0.0f);
            const int kB  = 992 + ((g >> 2) << 4) + (hh << 3) + ((g & 3) << 1);
            const int kBc = (kB < 998) ? kB : 998;
            const float2 tb = wrow2[kBc >> 1];
            b[2*g]   = (__bf16)((kB     < TWO_D) ? tb.x : 0.0f);
            b[2*g+1] = (__bf16)((kB + 1 < TWO_D) ? tb.y : 0.0f);
        }
        c = __builtin_amdgcn_wmma_f32_16x16x32_bf16(false, a, false, b,
                                                    (short)0, c, false, false);
    }
    // C/D 16x16 f32: VGPR r -> M = r + 8*hh ; N = lane%16
#pragma unroll
    for (int r = 0; r < 8; ++r)
        q_ws[(long long)(blockIdx.x * 16 + r + 8 * hh) * ATTN_DIM
             + blockIdx.y * 16 + m] = c[r];
}

// =====================================================================
// GEMM 2: v[B,1000] = q @ Wk   (bf16 WMMA); K=128 exact, N-edge via clamp
// =====================================================================
__global__ void __launch_bounds__(32)
vproj_wmma(const float* __restrict__ q_ws,
           const float* __restrict__ Wk,
           float* __restrict__ v_ws)
{
    const int lane = threadIdx.x;
    const int m  = lane & 15;
    const int hh = lane >> 4;

    const float2* arow2 = (const float2*)(q_ws + (long long)(blockIdx.x * 16 + m) * ATTN_DIM);
    const int n  = blockIdx.y * 16 + m;
    const int nc = (n < TWO_D) ? n : (TWO_D - 1);   // clamped load column

    v8f c = {};
#pragma unroll
    for (int k0 = 0; k0 < ATTN_DIM; k0 += 32) {
        v16bf a, b;
#pragma unroll
        for (int g = 0; g < 8; ++g) {
            const int kA = k0 + ((g >> 2) << 4) + (hh << 3) + ((g & 3) << 1);
            const float2 ta = arow2[kA >> 1];
            a[2*g]   = (__bf16)ta.x;
            a[2*g+1] = (__bf16)ta.y;
            const int kB = k0 + ((g >> 2) << 4) + (hh << 3) + ((g & 3) << 1);
            b[2*g]   = (__bf16)Wk[(long long)kB * TWO_D + nc];
            b[2*g+1] = (__bf16)Wk[(long long)(kB + 1) * TWO_D + nc];
        }
        c = __builtin_amdgcn_wmma_f32_16x16x32_bf16(false, a, false, b,
                                                    (short)0, c, false, false);
    }
#pragma unroll
    for (int r = 0; r < 8; ++r)
        if (n < TWO_D)
            v_ws[(long long)(blockIdx.x * 16 + r + 8 * hh) * TWO_D + n] = c[r];
}

// =====================================================================
// Main kernel: 1 block = 1 anchor, 256 threads (8 waves).
// LDS: hat bf16 (128KB) + e_i/v (8KB) + attn + per-wave double-buffered
// async staging of neighbor rows (64KB)  => ~202 KB of 320 KB WGP LDS
// =====================================================================
__global__ void __launch_bounds__(256)
refiner_main(const int* __restrict__ anchor_ids,
             const int* __restrict__ nbr_ent,
             const int* __restrict__ nbr_rel,
             const unsigned char* __restrict__ nbr_dir,
             const unsigned char* __restrict__ nbr_mask,
             const float* __restrict__ freq,
             const float* __restrict__ emb,
             const float* __restrict__ rel_phase,
             const float* __restrict__ a_param,
             const float* __restrict__ eta_raw,
             const float* __restrict__ w_raw,
             const float* __restrict__ b_p,
             const float* __restrict__ rel_bias,
             const float* __restrict__ dir_bias,
             const float* __restrict__ v_ws,
             float* __restrict__ out)
{
    extern __shared__ unsigned char smem[];
    __bf16* s_hat = (__bf16*)smem;                          // [64][1000] bf16
    float* s_ei  = (float*)(smem + OFF_EI);                 // [1000]
    float* s_v   = (float*)(smem + OFF_V);                  // [1000]
    float* s_at  = (float*)(smem + OFF_AT);                 // [64]
    int*   s_has = (int*)(smem + OFF_HAS);
    float* s_stage = (float*)(smem + OFF_STAGE);            // 8 waves * 2 * 1024 f32

    const int b      = blockIdx.x;
    const int tid    = threadIdx.x;
    const int anchor = anchor_ids[b];

    if (tid == 0) *s_has = 0;
    {
        const float* ei_row = emb  + (long long)anchor * TWO_D;
        const float* v_row  = v_ws + (long long)b * TWO_D;
        for (int d = tid; d < TWO_D; d += 256) { s_ei[d] = ei_row[d]; s_v[d] = v_row[d]; }
    }
    __syncthreads();

    const int lane = tid & 31;
    const int wv   = tid >> 5;

    float* cur = s_stage + wv * 2048;       // 1024 f32 per buffer
    float* nxt = cur + 1024;

    // prime the pipeline: async-copy this wave's first neighbor row into LDS
    issue_row(emb + (long long)nbr_ent[(long long)anchor * KN + wv] * TWO_D, cur, lane);

    // ---- pass 1: rotate neighbors, fused logit dot with v, stash hat in LDS ----
    for (int k = wv; k < KN; k += 8) {
        const long long nidx = (long long)anchor * KN + k;
        const int rel = nbr_rel[nidx];
        const int dir = nbr_dir[nidx] ? 1 : 0;
        const int msk = nbr_mask[nidx] ? 1 : 0;
        const float* ph = rel_phase + (long long)rel * DD;

        if (k + 8 < KN) {   // prefetch next neighbor row, then wait for current
            issue_row(emb + (long long)nbr_ent[nidx + 8] * TWO_D, nxt, lane);
            wait_async<8>();
        } else {
            wait_async<0>();
        }

        const float sgn = dir ? -1.0f : 1.0f;
        float part = 0.0f;
        for (int d = lane; d < DD; d += 32) {
            const float re_j = cur[d];
            const float im_j = cur[d + DD];
            const float p  = ph[d];
            const float cr = __cosf(p);
            const float sr = __sinf(p) * sgn;
            const float re_m = re_j * cr - im_j * sr;
            const float im_m = re_j * sr + im_j * cr;
            part += re_m * s_v[d] + im_m * s_v[d + DD];
            s_hat[k * TWO_D + d]      = (__bf16)re_m;
            s_hat[k * TWO_D + DD + d] = (__bf16)im_m;
        }
#pragma unroll
        for (int off = 16; off > 0; off >>= 1)
            part += __shfl_xor(part, off, 32);
        if (lane == 0) {
            float lg = part * INV_SQRT_ATTN + rel_bias[rel] + dir_bias[dir];
            if (!msk) lg = -10000.0f;
            else atomicOr(s_has, 1);
            s_at[k] = lg;
        }
        float* t = cur; cur = nxt; nxt = t;
    }
    __syncthreads();

    const int has = *s_has;

    // ---- softmax over K=64 (serial; trivial vs gather cost) ----
    if (tid == 0) {
        float mx = -1e30f;
        for (int k = 0; k < KN; ++k) mx = fmaxf(mx, s_at[k]);
        float sum = 0.0f;
        for (int k = 0; k < KN; ++k) sum += __expf(s_at[k] - mx);
        const float inv = has ? (1.0f / sum) : 0.0f;
        for (int k = 0; k < KN; ++k) s_at[k] = __expf(s_at[k] - mx) * inv;
    }
    __syncthreads();

    // ---- pass 2: delta = attn . hat, apply a_param and eta gate ----
    const float logfv = log1pf(freq[anchor]);
    const float wpar  = log1pf(__expf(w_raw[0]));            // softplus
    const float z     = eta_raw[0] - wpar * logfv + b_p[0];
    float eta = ETA_MAX / (1.0f + __expf(-z));
    if (!has) eta = 0.0f;                                    // -> out = e_i exactly

    float* orow = out + (long long)b * TWO_D;
    for (int d = tid; d < DD; d += 256) {
        float aR = 0.0f, aI = 0.0f;
#pragma unroll 4
        for (int k = 0; k < KN; ++k) {
            const float w = s_at[k];
            aR += w * (float)s_hat[k * TWO_D + d];
            aI += w * (float)s_hat[k * TWO_D + DD + d];
        }
        const float ap = a_param[d];
        const float dR = ap * aR;
        const float dI = ap * aI;
        const float eR = s_ei[d];
        const float eI = s_ei[d + DD];
        orow[d]      = eR + eta * (dR - eR);
        orow[d + DD] = eI + eta * (dI - eI);
    }
}

// =====================================================================
extern "C" void kernel_launch(void* const* d_in, const int* in_sizes, int n_in,
                              void* d_out, int out_size, void* d_ws, size_t ws_size,
                              hipStream_t stream) {
    const int*           anchor_ids = (const int*)d_in[0];
    const int*           nbr_ent    = (const int*)d_in[1];
    const int*           nbr_rel    = (const int*)d_in[2];
    const unsigned char* nbr_dir    = (const unsigned char*)d_in[3];
    const unsigned char* nbr_mask   = (const unsigned char*)d_in[4];
    const float*         freq       = (const float*)d_in[5];
    const float*         emb        = (const float*)d_in[6];
    const float*         rel_phase  = (const float*)d_in[7];
    const float*         a_param    = (const float*)d_in[8];
    const float*         eta_raw    = (const float*)d_in[9];
    const float*         w_raw      = (const float*)d_in[10];
    const float*         b_p        = (const float*)d_in[11];
    const float*         Wq         = (const float*)d_in[12];
    const float*         Wk         = (const float*)d_in[13];
    const float*         rel_bias   = (const float*)d_in[14];
    const float*         dir_bias   = (const float*)d_in[15];
    float*               out        = (float*)d_out;

    const int B = in_sizes[0];                           // 1024

    float* q_ws = (float*)d_ws;                          // [B,128]
    float* v_ws = q_ws + (size_t)B * ATTN_DIM;           // [B,1000]

    qproj_wmma<<<dim3(B / 16, ATTN_DIM / 16), 32, 0, stream>>>(anchor_ids, emb, Wq, q_ws);
    vproj_wmma<<<dim3(B / 16, (TWO_D + 15) / 16), 32, 0, stream>>>(q_ws, Wk, v_ws);

    refiner_main<<<B, 256, SMEM_TOTAL, stream>>>(anchor_ids, nbr_ent, nbr_rel, nbr_dir,
                                                 nbr_mask, freq, emb, rel_phase, a_param,
                                                 eta_raw, w_raw, b_p, rel_bias, dir_bias,
                                                 v_ws, out);
}
